// TinyDecoderLM_58162447123348
// MI455X (gfx1250) — matrix operand order repaired and checked by reading, not compiled, not run on hardware
//
#include <hip/hip_runtime.h>
#include <math.h>

// ---------------- problem constants ----------------
#define L_  8
#define D_  768
#define H_  12
#define HD_ 64
#define B_  2
#define S_  1024
#define V_  32000
#define DF_ 3072          // MR*D
#define T_  (B_*S_)       // total tokens = 2048

// ---------------- WMMA types / helpers ----------------
typedef __attribute__((ext_vector_type(16))) __bf16 v16bf;
typedef __attribute__((ext_vector_type(8)))  float  v8f;

struct alignas(16) U4 { unsigned int x, y, z, w; };

union Frag {
  v16bf v;
  U4 q[2];
  unsigned short us[16];
};

__device__ __forceinline__ unsigned short f2bf(float f) {
  unsigned u = __float_as_uint(f);
  unsigned r = (u + 0x7FFFu + ((u >> 16) & 1u)) >> 16;   // round-to-nearest-even
  return (unsigned short)r;
}

__device__ __forceinline__ float bf2f(unsigned short s) {
  return __uint_as_float(((unsigned)s) << 16);
}

#define WMMA_BF16(a, b, c) \
  __builtin_amdgcn_wmma_f32_16x16x32_bf16(false, (a), false, (b), (short)0, (c), false, false)

// A-operand (MxK tile, K=32): lane&15 = row M; lanes<16 hold K 0..15, lanes>=16 hold K 16..31 (contiguous)
// Works for both LDS tiles and global bf16 rows (stride in elements).
__device__ __forceinline__ v16bf fragA(const unsigned short* base, int row0, int stride, int k0, int lane) {
  int half = (lane >> 4) & 1;
  const unsigned short* p = base + (size_t)(row0 + (lane & 15)) * stride + k0 + half * 16;
  Frag f;
  f.q[0] = *(const U4*)(p);
  f.q[1] = *(const U4*)(p + 8);
  return f.v;
}

// B-operand (K=32 x N=16): lane&15 = output column; lanes<16 hold K 0..15, lanes>=16 hold K 16..31 (contiguous)
__device__ __forceinline__ v16bf fragB(const unsigned short* base, int row0, int stride, int k0, int lane) {
  int half = (lane >> 4) & 1;
  const unsigned short* p = base + (size_t)(row0 + (lane & 15)) * stride + k0 + half * 16;
  Frag f;
  f.q[0] = *(const U4*)(p);
  f.q[1] = *(const U4*)(p + 8);
  return f.v;
}

// ---------------- weight transpose + f32->bf16 convert ----------------
// W: (K x N) f32 row-major  ->  Wt: (N x K) bf16 row-major ; blockIdx.z = batch (layer)
__global__ __launch_bounds__(256) void tconv_kernel(const float* __restrict__ W,
                                                    unsigned short* __restrict__ Wt,
                                                    int K, int N) {
  W  += (size_t)blockIdx.z * K * N;
  Wt += (size_t)blockIdx.z * K * N;
  __shared__ float t[32][33];
  int tx = threadIdx.x & 31, ty = threadIdx.x >> 5;      // ty 0..7
  int k0 = blockIdx.y * 32, n0 = blockIdx.x * 32;
#pragma unroll
  for (int j = 0; j < 32; j += 8)
    t[ty + j][tx] = W[(size_t)(k0 + ty + j) * N + n0 + tx];
  __syncthreads();
#pragma unroll
  for (int j = 0; j < 32; j += 8)
    Wt[(size_t)(n0 + ty + j) * K + k0 + tx] = f2bf(t[tx][ty + j]);
}

// ---------------- embedding (f32 residual stream) ----------------
__global__ __launch_bounds__(256) void embed_kernel(const int* __restrict__ tokens,
                                                    const float* __restrict__ te,
                                                    const float* __restrict__ pe,
                                                    float* __restrict__ xo) {
  int row = blockIdx.x;             // b*S + s
  int s = row & (S_ - 1);
  int tok = tokens[row];
#pragma unroll
  for (int j = 0; j < 3; ++j) {
    int c = threadIdx.x + j * 256;
    xo[(size_t)row * D_ + c] = te[(size_t)tok * D_ + c] + pe[(size_t)s * D_ + c];
  }
}

// ---------------- layernorm: f32 in -> bf16 out (feeds WMMA A operand) ----------------
__global__ __launch_bounds__(256) void ln_kernel(const float* __restrict__ x,
                                                 const float* __restrict__ g,
                                                 const float* __restrict__ bb,
                                                 unsigned short* __restrict__ outp) {
  __shared__ float s1[256], s2[256];
  int row = blockIdx.x, tid = threadIdx.x;
  const float* xr = x + (size_t)row * D_;
  float sum = 0.f, sq = 0.f;
#pragma unroll
  for (int j = 0; j < 3; ++j) {
    float v = xr[tid + j * 256];
    sum += v; sq += v * v;
  }
  s1[tid] = sum; s2[tid] = sq;
  __syncthreads();
  for (int off = 128; off > 0; off >>= 1) {
    if (tid < off) { s1[tid] += s1[tid + off]; s2[tid] += s2[tid + off]; }
    __syncthreads();
  }
  float mean = s1[0] * (1.0f / D_);
  float var  = s2[0] * (1.0f / D_) - mean * mean;
  float rs = rsqrtf(var + 1e-5f);
#pragma unroll
  for (int j = 0; j < 3; ++j) {
    int c = tid + j * 256;
    outp[(size_t)row * D_ + c] = f2bf((xr[c] - mean) * rs * g[c] + bb[c]);
  }
}

// ---------------- bf16 WMMA GEMM ----------------
// A : (M x K) bf16 row-major (activations)
// Bt: (N x K) bf16 row-major (pre-transposed weight)
// Cout: f32 (OBF_=false) or bf16 (OBF_=true).
// GELU_: exact gelu on the result; RES_: += Res (f32, only with f32 out)
template <bool GELU_, bool RES_, bool OBF_>
__global__ __launch_bounds__(256) void gemm_kernel(const unsigned short* __restrict__ A,
                                                   const unsigned short* __restrict__ Bt,
                                                   const float* __restrict__ bias,
                                                   const float* __restrict__ Res,
                                                   void* __restrict__ Cout,
                                                   int M, int N, int K) {
  __shared__ __align__(16) unsigned short As[128 * 72];  // 128 x 64 (+8 pad)
  __shared__ __align__(16) unsigned short Bs[64 * 72];   // 64 x 64 (+8 pad)
  const int tid = threadIdx.x, lane = tid & 31, wave = tid >> 5;
  const int wm = (wave & 3) * 32, wn = (wave >> 2) * 32;
  const int m_base = blockIdx.y * 128, n_base = blockIdx.x * 64;

  v8f acc[2][2] = {};

  for (int kb = 0; kb < K; kb += 64) {
    __syncthreads();
#pragma unroll
    for (int it = 0; it < 4; ++it) {               // stage A tile (1024 x 16B)
      int idx = tid + it * 256;
      int row = idx >> 3, c8 = idx & 7;
      U4 val = *(const U4*)(A + (size_t)(m_base + row) * K + kb + c8 * 8);
      *(U4*)(&As[row * 72 + c8 * 8]) = val;
    }
#pragma unroll
    for (int it = 0; it < 2; ++it) {               // stage B tile (512 x 16B)
      int idx = tid + it * 256;
      int row = idx >> 3, c8 = idx & 7;
      U4 val = *(const U4*)(Bt + (size_t)(n_base + row) * K + kb + c8 * 8);
      *(U4*)(&Bs[row * 72 + c8 * 8]) = val;
    }
    if (kb + 64 < K) {                             // speculative prefetch of next tiles
      __builtin_prefetch(A + (size_t)(m_base + (tid >> 1)) * K + kb + 64, 0, 1);
      __builtin_prefetch(Bt + (size_t)(n_base + (tid & 63)) * K + kb + 64, 0, 1);
    }
    __syncthreads();
#pragma unroll
    for (int kk = 0; kk < 64; kk += 32) {
      v16bf a0 = fragA(As, wm + 0,  72, kk, lane);
      v16bf a1 = fragA(As, wm + 16, 72, kk, lane);
      v16bf b0 = fragB(Bs, wn + 0,  72, kk, lane);
      v16bf b1 = fragB(Bs, wn + 16, 72, kk, lane);
      acc[0][0] = WMMA_BF16(a0, b0, acc[0][0]);
      acc[0][1] = WMMA_BF16(a0, b1, acc[0][1]);
      acc[1][0] = WMMA_BF16(a1, b0, acc[1][0]);
      acc[1][1] = WMMA_BF16(a1, b1, acc[1][1]);
    }
  }

  const int rhalf = ((lane >> 4) & 1) * 8;
#pragma unroll
  for (int mt = 0; mt < 2; ++mt) {
#pragma unroll
    for (int nt = 0; nt < 2; ++nt) {
      int col = n_base + wn + nt * 16 + (lane & 15);
      float bv = bias[col];
#pragma unroll
      for (int e = 0; e < 8; ++e) {
        int row = m_base + wm + mt * 16 + e + rhalf;
        float v = acc[mt][nt][e] + bv;
        if (GELU_) v = 0.5f * v * (1.0f + erff(v * 0.70710678118654752f));
        if (RES_)  v += Res[(size_t)row * N + col];
        if (OBF_) ((unsigned short*)Cout)[(size_t)row * N + col] = f2bf(v);
        else      ((float*)Cout)[(size_t)row * N + col] = v;
      }
    }
  }
}

// ---------------- fused attention (token attention + layer-memory attention) ----------------
// All matrix operands are bf16 in global memory; ctx is written as bf16 (feeds Wo GEMM).
// One block per (q-tile of 32 rows, head, batch). Dynamic LDS:
//   sc  : f32  [32][1024]  score matrix
//   pb  : bf16 [32][1024]  normalized probabilities
//   vt  : bf16 [64][72]    transposed V chunk (64 HD x 64 keys, +8 pad)
//   msl : f32  [32][8]     memory-attention scores -> weights
//   red : f32  [32][8]     reduction scratch
#define ATTN_SMEM (32*1024*4 + 32*1024*2 + 64*72*2 + 32*8*4 + 32*8*4)

__global__ __launch_bounds__(256) void attn_kernel(const unsigned short* __restrict__ qrow,
                                                   const unsigned short* __restrict__ krow,
                                                   const unsigned short* __restrict__ vrow,
                                                   const unsigned short* __restrict__ qcolp,
                                                   const unsigned short* __restrict__ pastk,
                                                   const unsigned short* __restrict__ pastv,
                                                   unsigned short* __restrict__ ctxb,
                                                   int layer) {
  extern __shared__ char smem[];
  float*          sc  = (float*)smem;
  unsigned short* pb  = (unsigned short*)(smem + 32 * 1024 * 4);
  unsigned short* vt  = (unsigned short*)(smem + 32 * 1024 * 4 + 32 * 1024 * 2);
  float*          msl = (float*)(smem + 32 * 1024 * 4 + 32 * 1024 * 2 + 64 * 72 * 2);
  float*          red = msl + 32 * 8;

  const int b = blockIdx.z, h = blockIdx.y, q_base = blockIdx.x * 32;
  const int tid = threadIdx.x, lane = tid & 31, wave = tid >> 5;
  const int rhalf = ((lane >> 4) & 1) * 8;
  const float inv = 0.125f;                         // 1/sqrt(HD=64)

  const unsigned short* qbase = qrow + (size_t)b * S_ * D_ + h * HD_;
  const unsigned short* kbase = krow + (size_t)b * S_ * D_ + h * HD_;
  const unsigned short* vbase = vrow + (size_t)b * S_ * D_ + h * HD_;

  // ---- Q fragments straight from global bf16 (A operand) ----
  v16bf qf[2][2];
#pragma unroll
  for (int qt = 0; qt < 2; ++qt)
#pragma unroll
    for (int kk = 0; kk < 2; ++kk)
      qf[qt][kk] = fragA(qbase, q_base + qt * 16, D_, kk * 32, lane);

  // ---- phase 1: scores = Q K^T * inv with causal mask ----
  for (int kt = wave; kt < 64; kt += 8) {
    v16bf kf0 = fragB(kbase, kt * 16, D_, 0, lane);
    v16bf kf1 = fragB(kbase, kt * 16, D_, 32, lane);
#pragma unroll
    for (int qt = 0; qt < 2; ++qt) {
      v8f a = {};
      a = WMMA_BF16(qf[qt][0], kf0, a);
      a = WMMA_BF16(qf[qt][1], kf1, a);
#pragma unroll
      for (int e = 0; e < 8; ++e) {
        int rl = qt * 16 + e + rhalf;
        int ke = kt * 16 + (lane & 15);
        float v = a[e] * inv;
        if (ke > q_base + rl) v = -3.0e38f;         // causal mask (strict upper)
        sc[rl * 1024 + ke] = v;
      }
    }
  }

  // ---- layer-memory scores: q_col . past_k[l] at same position ----
  if (layer > 0) {
    int pairs = 32 * layer;
    for (int p = tid; p < pairs; p += 256) {
      int rl = p / layer, l = p % layer;
      int q = q_base + rl;
      const unsigned short* qp = qcolp + ((size_t)(b * S_ + q)) * D_ + h * HD_;
      const unsigned short* kp = pastk + (((size_t)l * B_ + b) * S_ + q) * D_ + h * HD_;
      float dot = 0.f;
#pragma unroll 8
      for (int d = 0; d < HD_; ++d) dot += bf2f(qp[d]) * bf2f(kp[d]);
      msl[rl * 8 + l] = dot * inv;
    }
  }
  __syncthreads();

  // ---- phase 2: joint softmax over [token scores, memory scores] ----
  {
    int rl = tid >> 3, t8 = tid & 7;
    float* srow = sc + rl * 1024;
    float pmax = -3.0e38f;
    for (int c = t8; c < 1024; c += 8) pmax = fmaxf(pmax, srow[c]);
    if (t8 == 0) for (int l = 0; l < layer; ++l) pmax = fmaxf(pmax, msl[rl * 8 + l]);
    red[rl * 8 + t8] = pmax;
    __syncthreads();
    float rmax = red[rl * 8];
#pragma unroll
    for (int j = 1; j < 8; ++j) rmax = fmaxf(rmax, red[rl * 8 + j]);
    __syncthreads();
    float psum = 0.f;
    for (int c = t8; c < 1024; c += 8) {
      float e = __expf(srow[c] - rmax);
      srow[c] = e; psum += e;
    }
    if (t8 == 0) for (int l = 0; l < layer; ++l) {
      float e = __expf(msl[rl * 8 + l] - rmax);
      msl[rl * 8 + l] = e; psum += e;
    }
    red[rl * 8 + t8] = psum;
    __syncthreads();
    float denom = 0.f;
#pragma unroll
    for (int j = 0; j < 8; ++j) denom += red[rl * 8 + j];
    float invd = 1.0f / denom;
    for (int c = t8; c < 1024; c += 8) pb[rl * 1024 + c] = f2bf(srow[c] * invd);
    if (t8 == 0) for (int l = 0; l < layer; ++l) msl[rl * 8 + l] *= invd;
  }

  // ---- phase 3: ctx = P @ V (+ memory weights @ past_v) ----
  const int qt = wave & 1, dt = wave >> 1;          // 2 q-tiles x 4 d-tiles across 8 waves
  v8f acc = {};
  for (int kc = 0; kc < 16; ++kc) {                 // 16 key-chunks of 64
    __syncthreads();
#pragma unroll
    for (int it = 0; it < 2; ++it) {                // stage transposed V chunk (bf16 -> bf16)
      int idx = tid + it * 256;                     // 0..511 x U4 (8 bf16 each)
      int key = idx >> 3;
      int d8  = (idx & 7) * 8;
      U4 u = *(const U4*)(vbase + ((size_t)(kc * 64 + key)) * D_ + d8);
      vt[(d8 + 0) * 72 + key] = (unsigned short)(u.x & 0xFFFFu);
      vt[(d8 + 1) * 72 + key] = (unsigned short)(u.x >> 16);
      vt[(d8 + 2) * 72 + key] = (unsigned short)(u.y & 0xFFFFu);
      vt[(d8 + 3) * 72 + key] = (unsigned short)(u.y >> 16);
      vt[(d8 + 4) * 72 + key] = (unsigned short)(u.z & 0xFFFFu);
      vt[(d8 + 5) * 72 + key] = (unsigned short)(u.z >> 16);
      vt[(d8 + 6) * 72 + key] = (unsigned short)(u.w & 0xFFFFu);
      vt[(d8 + 7) * 72 + key] = (unsigned short)(u.w >> 16);
    }
    __syncthreads();
#pragma unroll
    for (int kk = 0; kk < 64; kk += 32) {
      v16bf a  = fragA(pb, qt * 16, 1024, kc * 64 + kk, lane);
      v16bf bf = fragB(vt, dt * 16, 72, kk, lane);
      acc = WMMA_BF16(a, bf, acc);
    }
  }

  int d = dt * 16 + (lane & 15);
#pragma unroll
  for (int e = 0; e < 8; ++e) {
    int rl = qt * 16 + e + rhalf;
    int q = q_base + rl;
    float v = acc[e];
    for (int l = 0; l < layer; ++l)
      v += msl[rl * 8 + l] * bf2f(pastv[(((size_t)l * B_ + b) * S_ + q) * D_ + h * HD_ + d]);
    ctxb[((size_t)(b * S_ + q)) * D_ + h * HD_ + d] = f2bf(v);
  }
}

// ---------------- host orchestration ----------------
extern "C" void kernel_launch(void* const* d_in, const int* in_sizes, int n_in,
                              void* d_out, int out_size, void* d_ws, size_t ws_size,
                              hipStream_t stream) {
  (void)in_sizes; (void)n_in; (void)out_size; (void)ws_size;

  const int*   tokens  = (const int*)d_in[0];
  const float* tok_emb = (const float*)d_in[1];
  const float* pos_emb = (const float*)d_in[2];
  const float* Wq_row  = (const float*)d_in[3];
  const float* Wk_row  = (const float*)d_in[4];
  const float* Wv_row  = (const float*)d_in[5];
  const float* Wq_col  = (const float*)d_in[6];
  const float* Wo      = (const float*)d_in[7];
  const float* bq_row  = (const float*)d_in[8];
  const float* bk_row  = (const float*)d_in[9];
  const float* bv_row  = (const float*)d_in[10];
  const float* bq_col  = (const float*)d_in[11];
  const float* bo      = (const float*)d_in[12];
  const float* ln1_g   = (const float*)d_in[13];
  const float* ln1_b   = (const float*)d_in[14];
  const float* ln2_g   = (const float*)d_in[15];
  const float* ln2_b   = (const float*)d_in[16];
  const float* W1      = (const float*)d_in[17];
  const float* b1      = (const float*)d_in[18];
  const float* W2      = (const float*)d_in[19];
  const float* b2      = (const float*)d_in[20];
  const float* Wk_sh   = (const float*)d_in[21];
  const float* bk_sh   = (const float*)d_in[22];
  const float* Wv_sh   = (const float*)d_in[23];
  const float* bv_sh   = (const float*)d_in[24];
  const float* lnf_g   = (const float*)d_in[25];
  const float* lnf_b   = (const float*)d_in[26];
  const float* Whead   = (const float*)d_in[27];
  const float* bhead   = (const float*)d_in[28];
  float* out = (float*)d_out;

  // ---- workspace layout ----
  size_t off = 0;
  auto aus = [&](size_t n) { unsigned short* p = (unsigned short*)((char*)d_ws + off);
                             off = (off + n * 2 + 255) & ~(size_t)255; return p; };
  auto af  = [&](size_t n) { float* p = (float*)((char*)d_ws + off);
                             off = (off + n * 4 + 255) & ~(size_t)255; return p; };

  unsigned short* wqr  = aus((size_t)L_ * D_ * D_);
  unsigned short* wkr  = aus((size_t)L_ * D_ * D_);
  unsigned short* wvr  = aus((size_t)L_ * D_ * D_);
  unsigned short* wqc  = aus((size_t)L_ * D_ * D_);
  unsigned short* wot  = aus((size_t)L_ * D_ * D_);
  unsigned short* w1t  = aus((size_t)L_ * D_ * DF_);
  unsigned short* w2t  = aus((size_t)L_ * DF_ * D_);
  unsigned short* wksh = aus((size_t)D_ * D_);
  unsigned short* wvsh = aus((size_t)D_ * D_);
  unsigned short* wht  = aus((size_t)V_ * D_);
  unsigned short* abuf  = aus((size_t)T_ * DF_);   // A-operand buffer (LN out / ctx / lnf out)
  unsigned short* abuf2 = aus((size_t)T_ * DF_);   // GELU(mid) bf16 for W2
  unsigned short* qrow  = aus((size_t)T_ * D_);
  unsigned short* krow  = aus((size_t)T_ * D_);
  unsigned short* vrow  = aus((size_t)T_ * D_);
  unsigned short* qcol  = aus((size_t)T_ * D_);
  unsigned short* pastk = aus((size_t)L_ * T_ * D_);
  unsigned short* pastv = aus((size_t)L_ * T_ * D_);
  float* x = af((size_t)T_ * D_);                  // f32 residual stream

  hipFuncSetAttribute((const void*)attn_kernel,
                      hipFuncAttributeMaxDynamicSharedMemorySize, ATTN_SMEM);

  dim3 tb(256);

  // ---- weight transpose + bf16 conversion (once per launch) ----
  tconv_kernel<<<dim3(D_/32,  D_/32,  L_), tb, 0, stream>>>(Wq_row, wqr, D_, D_);
  tconv_kernel<<<dim3(D_/32,  D_/32,  L_), tb, 0, stream>>>(Wk_row, wkr, D_, D_);
  tconv_kernel<<<dim3(D_/32,  D_/32,  L_), tb, 0, stream>>>(Wv_row, wvr, D_, D_);
  tconv_kernel<<<dim3(D_/32,  D_/32,  L_), tb, 0, stream>>>(Wq_col, wqc, D_, D_);
  tconv_kernel<<<dim3(D_/32,  D_/32,  L_), tb, 0, stream>>>(Wo,     wot, D_, D_);
  tconv_kernel<<<dim3(DF_/32, D_/32,  L_), tb, 0, stream>>>(W1,     w1t, D_, DF_);
  tconv_kernel<<<dim3(D_/32,  DF_/32, L_), tb, 0, stream>>>(W2,     w2t, DF_, D_);
  tconv_kernel<<<dim3(D_/32,  D_/32,  1),  tb, 0, stream>>>(Wk_sh,  wksh, D_, D_);
  tconv_kernel<<<dim3(D_/32,  D_/32,  1),  tb, 0, stream>>>(Wv_sh,  wvsh, D_, D_);
  tconv_kernel<<<dim3(V_/32,  D_/32,  1),  tb, 0, stream>>>(Whead,  wht, D_, V_);

  // ---- embedding ----
  embed_kernel<<<T_, tb, 0, stream>>>(tokens, tok_emb, pos_emb, x);

  dim3 gD (D_ / 64,  T_ / 128);   // N=768
  dim3 gDF(DF_ / 64, T_ / 128);   // N=3072
  dim3 gV (V_ / 64,  T_ / 128);   // N=32000

  for (int i = 0; i < L_; ++i) {
    // abuf = bf16(LN1(x))
    ln_kernel<<<T_, tb, 0, stream>>>(x, ln1_g + i * D_, ln1_b + i * D_, abuf);

    // six D x D projections, bf16 outputs feed WMMA attention directly
    gemm_kernel<false, false, true><<<gD, tb, 0, stream>>>(abuf, wqr + (size_t)i * D_ * D_, bq_row + i * D_, nullptr, qrow, T_, D_, D_);
    gemm_kernel<false, false, true><<<gD, tb, 0, stream>>>(abuf, wkr + (size_t)i * D_ * D_, bk_row + i * D_, nullptr, krow, T_, D_, D_);
    gemm_kernel<false, false, true><<<gD, tb, 0, stream>>>(abuf, wvr + (size_t)i * D_ * D_, bv_row + i * D_, nullptr, vrow, T_, D_, D_);
    gemm_kernel<false, false, true><<<gD, tb, 0, stream>>>(abuf, wqc + (size_t)i * D_ * D_, bq_col + i * D_, nullptr, qcol, T_, D_, D_);
    gemm_kernel<false, false, true><<<gD, tb, 0, stream>>>(abuf, wksh, bk_sh, nullptr, pastk + (size_t)i * T_ * D_, T_, D_, D_);
    gemm_kernel<false, false, true><<<gD, tb, 0, stream>>>(abuf, wvsh, bv_sh, nullptr, pastv + (size_t)i * T_ * D_, T_, D_, D_);

    // fused causal + layer-memory attention; ctx (bf16) overwrites abuf (safe: stream order)
    attn_kernel<<<dim3(S_ / 32, H_, B_), tb, ATTN_SMEM, stream>>>(qrow, krow, vrow, qcol, pastk, pastv, abuf, i);

    // x = x + ctx @ Wo + bo
    gemm_kernel<false, true, false><<<gD, tb, 0, stream>>>(abuf, wot + (size_t)i * D_ * D_, bo + i * D_, x, x, T_, D_, D_);

    // MLP: x = x + gelu(LN2(x) @ W1 + b1) @ W2 + b2
    ln_kernel<<<T_, tb, 0, stream>>>(x, ln2_g + i * D_, ln2_b + i * D_, abuf);
    gemm_kernel<true, false, true><<<gDF, tb, 0, stream>>>(abuf, w1t + (size_t)i * D_ * DF_, b1 + i * DF_, nullptr, abuf2, T_, DF_, D_);
    gemm_kernel<false, true, false><<<gD, tb, 0, stream>>>(abuf2, w2t + (size_t)i * DF_ * D_, b2 + i * D_, x, x, T_, D_, DF_);
  }

  // final LN + LM head
  ln_kernel<<<T_, tb, 0, stream>>>(x, lnf_g, lnf_b, abuf);
  gemm_kernel<false, false, false><<<gV, tb, 0, stream>>>(abuf, wht, bhead, nullptr, out, T_, V_, D_);
}
